// VisualDepthAttention_32238024524164
// MI455X (gfx1250) — matrix operand order repaired and checked by reading, not compiled, not run on hardware
//
#include <hip/hip_runtime.h>

// ---------------------------------------------------------------------------
// VisualDepthAttention for MI455X (gfx1250): bf16 WMMA, f32 accumulate.
// One workgroup (256 threads = 8 wave32) per 64-token window; wave h owns head h.
// ---------------------------------------------------------------------------

typedef __attribute__((ext_vector_type(16))) __bf16       v16bf;
typedef __attribute__((ext_vector_type(8)))  __bf16       v8bf;
typedef __attribute__((ext_vector_type(2)))  __bf16       v2bf;
typedef __attribute__((ext_vector_type(8)))  float        v8f;
typedef __attribute__((ext_vector_type(4)))  unsigned int u32x4;

#define NTOK 64     // tokens per window
#define D1   257    // dim + 1
#define DM   256    // dim
#define KP   288    // padded K (9 chunks of 32)
#define NH   8      // heads
#define DH   32     // head dim
#define NOUTP 272   // padded output N (17 tiles of 16)
#define SCALE 0.17677669529663687f   // 32^-0.5

#if defined(__has_builtin) && __has_builtin(__builtin_amdgcn_sched_barrier)
#define SCHED_FENCE() __builtin_amdgcn_sched_barrier(0)
#else
#define SCHED_FENCE()
#endif

struct Frag2 { u32x4 lo, hi; };

static __device__ __forceinline__ v16bf frag_cast(u32x4 lo, u32x4 hi) {
  Frag2 t{lo, hi};
  return __builtin_bit_cast(v16bf, t);
}

// A fragment: 16x32 bf16 tile from row-major storage (leading dim ldm elems).
// Lane layout (ISA 7.12.2): lane&15 = row, elements i<8 -> K=kb+8*h2+i,
// elements i>=8 -> K=kb+16+8*h2+(i-8).  -> two contiguous 16B loads.
static __device__ __forceinline__ v16bf load_a(const __bf16* base, int ldm,
                                               int m0, int kb, int lane) {
  const int h2 = lane >> 4;
  const int m  = m0 + (lane & 15);
  const __bf16* p = base + m * ldm + kb + 8 * h2;
  u32x4 lo = *(const u32x4*)p;
  u32x4 hi = *(const u32x4*)(p + 16);
  return frag_cast(lo, hi);
}

// B fragment: 32x16 bf16 tile, element (k,n) stored at base[n*ldk + k]
// (N-major).  Lane layout: lane&15 = n, i<8 -> K=kb+8*h2+i, i>=8 -> K=kb+16+8*h2+(i-8).
static __device__ __forceinline__ v16bf load_b(const __bf16* base, int ldk,
                                               int n0, int kb, int lane) {
  const int h2 = lane >> 4;
  const int n  = n0 + (lane & 15);
  const __bf16* p = base + n * ldk + kb + 8 * h2;
  u32x4 lo = *(const u32x4*)p;
  u32x4 hi = *(const u32x4*)(p + 16);
  return frag_cast(lo, hi);
}

static __device__ __forceinline__ v8f wmma_bf16(v16bf a, v16bf b, v8f c) {
  return __builtin_amdgcn_wmma_f32_16x16x32_bf16(false, a, false, b,
                                                 (short)0, c, false, false);
}

static __device__ __forceinline__ float sigmoidf_(float x) {
  return 1.0f / (1.0f + __expf(-x));
}

// ---------------------------------------------------------------------------
// Weight prep: fp32 -> bf16, transpose to N-major, zero-pad K 257->288.
// ---------------------------------------------------------------------------
__global__ void vda_prep_weights(const float* __restrict__ Wqv,
                                 const float* __restrict__ Wkv,
                                 const float* __restrict__ Wvv,
                                 const float* __restrict__ Wout,
                                 __bf16* __restrict__ wq,
                                 __bf16* __restrict__ wk,
                                 __bf16* __restrict__ wv,
                                 __bf16* __restrict__ wo) {
  const int idx = blockIdx.x * blockDim.x + threadIdx.x;
  if (idx < DM * KP) {
    const int n = idx / KP, k = idx % KP;
    const bool ok = (k < D1);
    wq[idx] = (__bf16)(ok ? Wqv[k * DM + n] : 0.0f);
    wk[idx] = (__bf16)(ok ? Wkv[k * DM + n] : 0.0f);
    wv[idx] = (__bf16)(ok ? Wvv[k * DM + n] : 0.0f);
  }
  if (idx < NOUTP * KP) {
    const int n = idx / KP, k = idx % KP;
    wo[idx] = (__bf16)((k < D1 && n < D1) ? Wout[k * D1 + n] : 0.0f);
  }
}

// One K-chunk of a 64-row GEMM: batch-issue 4 A-frags + 1 B-frag, fence,
// then 4 back-to-back WMMAs -> one dscnt wait per chunk instead of four.
static __device__ __forceinline__ void gemm_chunk(const __bf16* sA, int lda,
                                                  const __bf16* W, int ldb,
                                                  int nt16, int kb, int lane,
                                                  v8f acc[4]) {
  const v16bf b  = load_b(W, ldb, nt16, kb, lane);
  const v16bf a0 = load_a(sA, lda,  0, kb, lane);
  const v16bf a1 = load_a(sA, lda, 16, kb, lane);
  const v16bf a2 = load_a(sA, lda, 32, kb, lane);
  const v16bf a3 = load_a(sA, lda, 48, kb, lane);
  SCHED_FENCE();
  acc[0] = wmma_bf16(a0, b, acc[0]);
  acc[1] = wmma_bf16(a1, b, acc[1]);
  acc[2] = wmma_bf16(a2, b, acc[2]);
  acc[3] = wmma_bf16(a3, b, acc[3]);
  SCHED_FENCE();
}

// ---------------------------------------------------------------------------
// Compile-time specialized projection GEMM.  Per output column: 4 M-tile
// accumulators, one B-fragment per K-chunk reused across the 4 WMMAs.
// WHICH: 0 -> sQ, 1 -> sK (row-major), 2 -> sVt (transposed, packed 16B
// ds_store_b128: rows contiguous per lane in the transposed layout).
// ---------------------------------------------------------------------------
template <int WHICH>
static __device__ __forceinline__ void proj_gemm(const __bf16* __restrict__ W,
                                                 const __bf16* sX,
                                                 __bf16* sQ, __bf16* sK,
                                                 __bf16* sVt,
                                                 int wid, int lane,
                                                 int n_lane, int h2) {
#pragma unroll
  for (int nn = 0; nn < 2; ++nn) {
    const int nt = wid * 2 + nn;
    const int n  = nt * 16 + n_lane;
    v8f acc[4];
#pragma unroll
    for (int mt = 0; mt < 4; ++mt) acc[mt] = (v8f){};
#pragma unroll
    for (int kc = 0; kc < 9; ++kc)
      gemm_chunk(sX, KP, W, KP, nt * 16, kc * 32, lane, acc);
#pragma unroll
    for (int mt = 0; mt < 4; ++mt) {
      if (WHICH == 2) {
        v8bf pk;
#pragma unroll
        for (int r = 0; r < 8; ++r) pk[r] = (__bf16)acc[mt][r];
        *(v8bf*)&sVt[n * NTOK + mt * 16 + 8 * h2] = pk;   // ds_store_b128
      } else {
        __bf16* dst = (WHICH == 0) ? sQ : sK;
#pragma unroll
        for (int r = 0; r < 8; ++r)
          dst[(mt * 16 + r + 8 * h2) * DM + n] = (__bf16)acc[mt][r];
      }
    }
  }
}

// ---------------------------------------------------------------------------
// Main kernel. Dynamic LDS layout (bytes):
//   sX / sCat : [64][288] bf16      @      0   (36864)  -- x, reused as concat
//   sQ        : [64][256] bf16      @  36864   (32768)
//   sK        : [64][256] bf16      @  69632   (32768)
//   sVt       : [256][64] bf16      @ 102400   (32768)  -- V transposed
//   sAttn     : [8][64][64] bf16    @ 135168   (65536)
// total dynamic = 200704 B; static: sSum 16KB + depth vectors.
// ---------------------------------------------------------------------------
#define SMEM_BYTES 200704

__global__ void vda_attn_kernel(const float* __restrict__ x,
                                const float* __restrict__ Wqd,
                                const float* __restrict__ Wkd,
                                const float* __restrict__ Wvd,
                                const __bf16* __restrict__ wq,
                                const __bf16* __restrict__ wk,
                                const __bf16* __restrict__ wv,
                                const __bf16* __restrict__ wo,
                                float* __restrict__ out) {
  extern __shared__ char smem[];
  __bf16* sX    = (__bf16*)smem;                 // 64 x 288, reused as sCat
  __bf16* sQ    = (__bf16*)(smem + 36864);       // 64 x 256
  __bf16* sK    = (__bf16*)(smem + 69632);       // 64 x 256
  __bf16* sVt   = (__bf16*)(smem + 102400);      // 256 x 64 (dim-major)
  __bf16* sAttn = (__bf16*)(smem + 135168);      // 8 x 64 x 64
  __bf16* sCat  = sX;

  __shared__ float sSum[NTOK * NTOK];            // cross-head gated-score sum
  __shared__ float sQd[NTOK], sKd[NTOK], sVd[NTOK];

  const int tid    = threadIdx.x;
  const int lane   = tid & 31;
  const int wid    = tid >> 5;                   // 8 waves == 8 heads
  const int n_lane = lane & 15;
  const int h2     = lane >> 4;
  const int blk    = blockIdx.x;
  const long xbase = (long)blk * NTOK * D1;

  // ---- Phase 1: stage x into LDS as bf16 (zero-padded), packed pair stores.
  // Global loads are scalar b32 (row base only 4B-aligned: stride 257 floats).
  for (int i = wid; i < NTOK; i += 8) {
    const float* xrow = x + xbase + (long)i * D1;
    for (int p = lane; p < KP / 2; p += 32) {
      const int k = 2 * p;
      const float a = (k < D1)     ? xrow[k]     : 0.0f;
      const float b = (k + 1 < D1) ? xrow[k + 1] : 0.0f;
      v2bf w = { (__bf16)a, (__bf16)b };
      *(v2bf*)&sX[i * KP + k] = w;               // ds_store_b32
    }
  }
  for (int t = tid; t < NTOK * NTOK; t += 256) sSum[t] = 0.0f;
  __syncthreads();

  // ---- Phase 2: Q/K/V projections (64x288 @ 288x256) with bf16 WMMA.
  proj_gemm<0>(wq, sX, sQ, sK, sVt, wid, lane, n_lane, h2);
  proj_gemm<1>(wk, sX, sQ, sK, sVt, wid, lane, n_lane, h2);
  proj_gemm<2>(wv, sX, sQ, sK, sVt, wid, lane, n_lane, h2);

  // depth-path projections: 4 lanes per token row, shfl-reduced
  {
    const int p = tid & 3, i = tid >> 2;         // i in [0,64)
    float aq = 0.0f, ak = 0.0f, av = 0.0f;
    for (int k = p; k < D1; k += 4) {
      const float xv = (float)sX[i * KP + k];
      aq += xv * Wqd[k];
      ak += xv * Wkd[k];
      av += xv * Wvd[k];
    }
#pragma unroll
    for (int mk = 1; mk < 4; mk <<= 1) {
      aq += __shfl_xor(aq, mk, 32);
      ak += __shfl_xor(ak, mk, 32);
      av += __shfl_xor(av, mk, 32);
    }
    if (p == 0) { sQd[i] = aq * SCALE; sKd[i] = ak; sVd[i] = av; }
  }
  __syncthreads();

  // ---- Phase 3: per-head scores, depth gating, cross-head sum, softmax.
  const int h = wid;
  v8f acc[4][4];
#pragma unroll
  for (int mt = 0; mt < 4; ++mt)
#pragma unroll
    for (int nt = 0; nt < 4; ++nt) acc[mt][nt] = (v8f){};

#pragma unroll
  for (int mt = 0; mt < 4; ++mt) {
    v16bf a = load_a(sQ, DM, mt * 16, h * DH, lane);   // q rows, K = head dims
#pragma unroll
    for (int nt = 0; nt < 4; ++nt) {
      v16bf b = load_b(sK, DM, nt * 16, h * DH, lane); // B[d,j] = k[j,d]
      acc[mt][nt] = wmma_bf16(a, b, acc[mt][nt]);
    }
  }

  float kdv[4];
#pragma unroll
  for (int nt = 0; nt < 4; ++nt) kdv[nt] = sKd[nt * 16 + n_lane];

  // fused gating + cross-head LDS atomic sum + row softmax per (mt, r)
#pragma unroll
  for (int mt = 0; mt < 4; ++mt)
#pragma unroll
    for (int r = 0; r < 8; ++r) {
      const int i = mt * 16 + r + 8 * h2;
      const float qg = sQd[i];
#pragma unroll
      for (int nt = 0; nt < 4; ++nt) {
        const int j = nt * 16 + n_lane;
        const float s = acc[mt][nt][r] * SCALE * sigmoidf_(qg * kdv[nt]);
        acc[mt][nt][r] = s;
        atomicAdd(&sSum[i * NTOK + j], s);       // ds_add_f32
      }
      float m = acc[mt][0][r];
#pragma unroll
      for (int nt = 1; nt < 4; ++nt) m = fmaxf(m, acc[mt][nt][r]);
#pragma unroll
      for (int mk = 1; mk < 16; mk <<= 1) m = fmaxf(m, __shfl_xor(m, mk, 32));
      float sum = 0.0f;
#pragma unroll
      for (int nt = 0; nt < 4; ++nt) {
        const float e = __expf(acc[mt][nt][r] - m);
        acc[mt][nt][r] = e;
        sum += e;
      }
#pragma unroll
      for (int mk = 1; mk < 16; mk <<= 1) sum += __shfl_xor(sum, mk, 32);
      const float inv = 1.0f / sum;
#pragma unroll
      for (int nt = 0; nt < 4; ++nt)
        sAttn[h * (NTOK * NTOK) + i * NTOK + nt * 16 + n_lane] =
            (__bf16)(acc[mt][nt][r] * inv);
    }
  __syncthreads();

  // ---- Phase 4a: out_v = attn @ V per head -> concat buffer (cols 0..255)
  const __bf16* sAh = sAttn + h * (NTOK * NTOK);
#pragma unroll
  for (int nt = 0; nt < 2; ++nt) {
    v16bf b0 = load_b(sVt, NTOK, h * DH + nt * 16, 0, lane);
    v16bf b1 = load_b(sVt, NTOK, h * DH + nt * 16, 32, lane);
    const int d = h * DH + nt * 16 + n_lane;
#pragma unroll
    for (int mt = 0; mt < 4; ++mt) {
      v8f oc = {};
      oc = wmma_bf16(load_a(sAh, NTOK, mt * 16, 0, lane), b0, oc);
      oc = wmma_bf16(load_a(sAh, NTOK, mt * 16, 32, lane), b1, oc);
#pragma unroll
      for (int r = 0; r < 8; ++r)
        sCat[(mt * 16 + r + 8 * h2) * KP + d] = (__bf16)oc[r];
    }
  }

  // ---- Phase 4b: depth attention (softmax of head-summed scores) -> out_d
  {
    const int p = tid & 3, i = tid >> 2;
    float m = -3.0e38f;
    for (int j = p; j < NTOK; j += 4) m = fmaxf(m, sSum[i * NTOK + j]);
#pragma unroll
    for (int mk = 1; mk < 4; mk <<= 1) m = fmaxf(m, __shfl_xor(m, mk, 32));
    float sum = 0.0f, od = 0.0f;
    for (int j = p; j < NTOK; j += 4) {
      const float e = __expf(sSum[i * NTOK + j] - m);
      sum += e;
      od += e * sVd[j];
    }
#pragma unroll
    for (int mk = 1; mk < 4; mk <<= 1) {
      sum += __shfl_xor(sum, mk, 32);
      od  += __shfl_xor(od, mk, 32);
    }
    if (p == 0) sCat[i * KP + 256] = (__bf16)(od / sum);
  }
  // zero pad columns 257..287 of the concat buffer
  for (int t = tid; t < NTOK * 32; t += 256) {
    const int i = t >> 5, c = DM + (t & 31);
    if (c >= D1) sCat[i * KP + c] = (__bf16)0.0f;
  }
  __syncthreads();

  // ---- Phase 5: out = concat(out_v, out_d) @ Wout  (64x288 @ 288x272)
  for (int nt = wid; nt < 17; nt += 8) {
    const int n = nt * 16 + n_lane;
    v8f oc[4];
#pragma unroll
    for (int mt = 0; mt < 4; ++mt) oc[mt] = (v8f){};
#pragma unroll
    for (int kc = 0; kc < 9; ++kc)
      gemm_chunk(sCat, KP, wo, KP, nt * 16, kc * 32, lane, oc);
    if (n < D1) {
#pragma unroll
      for (int mt = 0; mt < 4; ++mt)
#pragma unroll
        for (int r = 0; r < 8; ++r)
          out[((long)blk * NTOK + mt * 16 + r + 8 * h2) * D1 + n] = oc[mt][r];
    }
  }
}

// ---------------------------------------------------------------------------
extern "C" void kernel_launch(void* const* d_in, const int* in_sizes, int n_in,
                              void* d_out, int out_size, void* d_ws, size_t ws_size,
                              hipStream_t stream) {
  const float* x    = (const float*)d_in[0];
  const float* Wqv  = (const float*)d_in[1];
  const float* Wkv  = (const float*)d_in[2];
  const float* Wvv  = (const float*)d_in[3];
  const float* Wqd  = (const float*)d_in[4];
  const float* Wkd  = (const float*)d_in[5];
  const float* Wvd  = (const float*)d_in[6];
  const float* Wout = (const float*)d_in[7];
  float* out = (float*)d_out;

  // d_ws layout: bf16 padded N-major weights
  __bf16* wq = (__bf16*)d_ws;            // 256*288
  __bf16* wk = wq + DM * KP;
  __bf16* wv = wk + DM * KP;
  __bf16* wo = wv + DM * KP;             // 272*288

  const int nwin = in_sizes[0] / (NTOK * D1);    // 2048

  hipFuncSetAttribute((const void*)vda_attn_kernel,
                      hipFuncAttributeMaxDynamicSharedMemorySize, SMEM_BYTES);

  vda_prep_weights<<<(NOUTP * KP + 255) / 256, 256, 0, stream>>>(
      Wqv, Wkv, Wvv, Wout, wq, wk, wv, wo);

  vda_attn_kernel<<<nwin, 256, SMEM_BYTES, stream>>>(
      x, Wqd, Wkd, Wvd, wq, wk, wv, wo, out);
}
